// MultilayersGAT_48430051229944
// MI455X (gfx1250) — matrix operand order repaired and checked by reading, not compiled, not run on hardware
//
#include <hip/hip_runtime.h>
#include <cstdint>
#include <cstddef>

#define N_NODES 1024
#define EF 32
#define ALPHA 0.2f
#define NEGV -9.0e15f

typedef _Float16 v16h __attribute__((ext_vector_type(16)));
typedef _Float16 v8h  __attribute__((ext_vector_type(8)));
typedef float    v8f  __attribute__((ext_vector_type(8)));
typedef int      v4i_vs __attribute__((vector_size(16)));   // matches builtin param pointee

union HFrag { v16h v; v8h h8[2]; };

// ---------- async global->LDS copy path (guarded; falls back to sync copy) ----------
#if defined(__HIP_DEVICE_COMPILE__) && defined(__has_builtin)
#if __has_builtin(__builtin_amdgcn_global_load_async_to_lds_b128) && \
    __has_builtin(__builtin_amdgcn_s_wait_asynccnt)
#define ASYNC_LDS 1
#endif
#endif

__device__ __forceinline__ void cp16_to_lds(const _Float16* g, _Float16* l) {
#ifdef ASYNC_LDS
    __builtin_amdgcn_global_load_async_to_lds_b128(
        (__attribute__((address_space(1))) v4i_vs*)g,
        (__attribute__((address_space(3))) v4i_vs*)l, 0, 0);
#else
    *(v8h*)l = *(const v8h*)g;     // fallback: global_load_b128 + ds_store_b128
#endif
}

template <int N>
__device__ __forceinline__ void wait_async() {
#ifdef ASYNC_LDS
    __builtin_amdgcn_s_wait_asynccnt(N);
#endif
}

// ---- copy x into output cols [0,256) and make f16 copy for layer-1 GEMM ----
__global__ void init_kernel(const float* __restrict__ x, float* __restrict__ out,
                            _Float16* __restrict__ cur) {
    int t = blockIdx.x * 256 + threadIdx.x;        // 1024*256 threads
    int i = t >> 8, c = t & 255;
    float v = x[t];
    out[(size_t)i * 768 + c] = v;
    cur[t] = (_Float16)v;
}

// ---- c_edge[l,h,:] = Wk_edge[l,h] @ a2[l,h]  (8 combos x 32 feats) ----
__global__ void cedge_kernel(const float* __restrict__ Wk, const float* __restrict__ a,
                             float* __restrict__ cedge) {
    int t = threadIdx.x;                           // 256 threads, 1 block
    int lh = t >> 5, f = t & 31;
    const float* wrow = Wk + ((size_t)lh * 288 + 256 + f) * 64;
    const float* a2   = a + (size_t)lh * 128 + 64;
    float s = 0.f;
    #pragma unroll 8
    for (int d = 0; d < 64; ++d) s += wrow[d] * a2[d];
    cedge[t] = s;
}

// ---- se[lh][i][j] = e[i,j,:] . c_edge[lh]  — single streaming pass over e ----
__global__ void edge_scores_kernel(const float* __restrict__ e, const float* __restrict__ cedge,
                                   float* __restrict__ se) {
    __shared__ float sc[256];
    int tid = threadIdx.x;
    sc[tid] = cedge[tid];
    __syncthreads();
    size_t t = (size_t)blockIdx.x * 256 + tid;     // one (i,j) per thread
    const float* ep = e + t * EF;
    __builtin_prefetch(ep + 256 * EF, 0, 0);       // global_prefetch_b8
    float ev[EF];
    #pragma unroll
    for (int f = 0; f < EF; ++f) ev[f] = ep[f];
    #pragma unroll
    for (int co = 0; co < 8; ++co) {
        float s = 0.f;
        #pragma unroll
        for (int f = 0; f < EF; ++f) s += ev[f] * sc[co * EF + f];
        se[(size_t)co * ((size_t)N_NODES * N_NODES) + t] = s;
    }
}

// ---- pack per-layer weights into f16 B^T [768][256]: per head [Wq | Wk_node | Wv_node] ----
__global__ void pack_w_kernel(const float* __restrict__ Wq, const float* __restrict__ Wk,
                              const float* __restrict__ Wv, int l, _Float16* __restrict__ wt) {
    int t = blockIdx.x * 256 + threadIdx.x;        // 768*256 threads
    int c = t >> 8, k = t & 255;
    int h = c / 192, r = c % 192;
    int sel = r >> 6, d = r & 63;
    float v;
    if (sel == 0)      v = Wq[(((size_t)(l * 4 + h)) * 256 + k) * 64 + d];
    else if (sel == 1) v = Wk[(((size_t)(l * 4 + h)) * 288 + k) * 64 + d];
    else               v = Wv[(((size_t)(l * 4 + h)) * 288 + k) * 64 + d];
    wt[(size_t)c * 256 + k] = (_Float16)v;
}

// ---- batched WMMA GEMM with double-buffered async LDS staging ----
// C[M,N] f32 = A[M,K] f16 (row-major) x Bt[N,K] f16.  Block: 32x64 tile, 8 waves,
// each wave one 16x16 WMMA tile.  Per K-step: A panel 32x32, B panel 64x32 staged
// via global_load_async_to_lds_b128 (ASYNCcnt) and consumed with ds_load_b128.
__global__ void wmma_gemm_kernel(const _Float16* __restrict__ A, const _Float16* __restrict__ Bt,
                                 float* __restrict__ C, int lda, int ldbt, int ldc, int K,
                                 size_t strideA, size_t strideB, size_t strideC) {
    __shared__ _Float16 abuf[2][32 * 32];          // 2 x 2KB
    __shared__ _Float16 bbuf[2][64 * 32];          // 2 x 4KB
    int t      = threadIdx.x;
    int lane   = t & 31;
    int wave   = t >> 5;                           // 0..7
    int laneLo = lane & 15, laneHi = lane >> 4;
    int mBase  = (wave >> 2) * 16;                 // 0 or 16
    int nBase  = (wave & 3) * 16;                  // 0,16,32,48
    const _Float16* Abase = A + strideA * blockIdx.z + (size_t)(blockIdx.y * 32) * lda;
    const _Float16* Bbase = Bt + strideB * blockIdx.z + (size_t)(blockIdx.x * 64) * ldbt;

    // per-step staging: A: 128 b128 loads (threads 0..127), B: 256 b128 loads (all threads)
    int rA = t >> 2, segA = t & 3;                 // rA 0..63 (A uses rA<32 via t<128)
    auto issue = [&](int k0, int b) {
        if (t < 128)
            cp16_to_lds(Abase + (size_t)rA * lda + k0 + segA * 8, &abuf[b][rA * 32 + segA * 8]);
        cp16_to_lds(Bbase + (size_t)rA * ldbt + k0 + segA * 8, &bbuf[b][rA * 32 + segA * 8]);
    };

    v8f acc = {};
    int S = K >> 5;
    issue(0, 0);
    for (int s = 0; s < S; ++s) {
        int cur = s & 1;
        if (s + 1 < S) {
            issue((s + 1) << 5, cur ^ 1);
            if (t < 128) wait_async<2>(); else wait_async<1>();  // current panel complete
        } else {
            wait_async<0>();
        }
        __syncthreads();                           // panel visible to all waves
        HFrag af, bf;
        const _Float16* As = &abuf[cur][0];
        const _Float16* Bs = &bbuf[cur][0];
        af.h8[0] = *(const v8h*)(As + (mBase + laneLo) * 32 + laneHi * 8);
        af.h8[1] = *(const v8h*)(As + (mBase + laneLo) * 32 + laneHi * 8 + 16);
        bf.h8[0] = *(const v8h*)(Bs + (nBase + laneLo) * 32 + laneHi * 8);
        bf.h8[1] = *(const v8h*)(Bs + (nBase + laneLo) * 32 + laneHi * 8 + 16);
        acc = __builtin_amdgcn_wmma_f32_16x16x32_f16(false, af.v, false, bf.v,
                                                     (short)0, acc, false, false);
        __syncthreads();                           // all reads done before buffer reuse
    }
    float* Cp = C + strideC * blockIdx.z;
    int tileM = blockIdx.y * 32 + mBase;
    int tileN = blockIdx.x * 64 + nBase;
    #pragma unroll
    for (int r = 0; r < 8; ++r)
        Cp[(size_t)(tileM + laneHi * 8 + r) * ldc + tileN + laneLo] = acc[r];
}

// ---- from qkv: sq[i,h]=hq.a1, sk[i,h]=kn.a2, and transpose vn to f16 [h][64][1024] ----
__global__ void qkv_post_kernel(const float* __restrict__ qkv, const float* __restrict__ a, int l,
                                float* __restrict__ sq, float* __restrict__ sk,
                                _Float16* __restrict__ vnT) {
    int t = blockIdx.x * 256 + threadIdx.x;        // 4096 threads
    int i = t >> 2, h = t & 3;
    const float* q  = qkv + (size_t)i * 768 + h * 192;
    const float* av = a + (size_t)(l * 4 + h) * 128;
    float s1 = 0.f, s2 = 0.f;
    #pragma unroll 8
    for (int d = 0; d < 64; ++d) { s1 += q[d] * av[d]; s2 += q[64 + d] * av[64 + d]; }
    sq[t] = s1; sk[t] = s2;
    for (int d = 0; d < 64; ++d)
        vnT[((size_t)h * 64 + d) * N_NODES + i] = (_Float16)q[128 + d];
}

// ---- fused: scores -> leaky_relu -> mask -> softmax (unnormalized) -> att f16,
//      plus attention-weighted reduction of e (32-wide). one block per (i, head). ----
__global__ void attn_kernel(const float* __restrict__ se, const int* __restrict__ adj,
                            const float* __restrict__ e, const float* __restrict__ sq,
                            const float* __restrict__ sk, int l,
                            _Float16* __restrict__ att, float* __restrict__ denom,
                            float* __restrict__ ae) {
    int i = blockIdx.x, h = blockIdx.y, tid = threadIdx.x;
    const float* serow = se + ((size_t)(l * 4 + h)) * N_NODES * (size_t)N_NODES + (size_t)i * N_NODES;
    const int* adjrow  = adj + (size_t)i * N_NODES;
    const float* erow  = e + (size_t)i * N_NODES * EF;
    float q = sq[i * 4 + h];
    __shared__ float red[256];
    __shared__ float s_ae[EF];
    // pass 1: row max
    float mx = -3.0e38f;
    for (int j = tid; j < N_NODES; j += 256) {
        float s = q + sk[j * 4 + h] + serow[j];
        s = s > 0.f ? s : ALPHA * s;
        s = adjrow[j] > 0 ? s : NEGV;
        mx = fmaxf(mx, s);
    }
    red[tid] = mx; __syncthreads();
    for (int st = 128; st > 0; st >>= 1) { if (tid < st) red[tid] = fmaxf(red[tid], red[tid + st]); __syncthreads(); }
    mx = red[0];
    __syncthreads();
    if (tid < EF) s_ae[tid] = 0.f;
    float aereg[EF];
    #pragma unroll
    for (int f = 0; f < EF; ++f) aereg[f] = 0.f;
    float den = 0.f;
    _Float16* attrow = att + (size_t)h * N_NODES * (size_t)N_NODES + (size_t)i * N_NODES;
    __syncthreads();
    // pass 2: exp, denom, weighted e-aggregation, write unnormalized att
    for (int j = tid; j < N_NODES; j += 256) {
        float s = q + sk[j * 4 + h] + serow[j];
        s = s > 0.f ? s : ALPHA * s;
        s = adjrow[j] > 0 ? s : NEGV;
        float p = __expf(s - mx);
        attrow[j] = (_Float16)p;
        den += p;
        const float* ej = erow + (size_t)j * EF;
        if (j + 256 < N_NODES) __builtin_prefetch(ej + 256 * EF, 0, 0);
        if (p > 0.f) {
            #pragma unroll
            for (int f = 0; f < EF; ++f) aereg[f] += p * ej[f];
        }
    }
    red[tid] = den; __syncthreads();
    for (int st = 128; st > 0; st >>= 1) { if (tid < st) red[tid] += red[tid + st]; __syncthreads(); }
    if (tid == 0) denom[i * 4 + h] = red[0];
    #pragma unroll
    for (int f = 0; f < EF; ++f) atomicAdd(&s_ae[f], aereg[f]);
    __syncthreads();
    if (tid < EF) ae[((size_t)i * 4 + h) * EF + tid] = s_ae[tid];
}

// ---- finalize: out = elu((att@vn + ae@Wv_edge)/denom); write fp32 out + f16 next-layer input ----
__global__ void finalize_kernel(const float* __restrict__ nodeagg, const float* __restrict__ ae,
                                const float* __restrict__ denom, const float* __restrict__ Wv,
                                int l, float* __restrict__ out, _Float16* __restrict__ cur) {
    int t = blockIdx.x * 256 + threadIdx.x;        // 1024*256 threads
    int i = t >> 8, c = t & 255;
    int h = c >> 6, d = c & 63;
    const float* aerow = ae + ((size_t)i * 4 + h) * EF;
    const float* wv = Wv + (((size_t)(l * 4 + h)) * 288 + 256) * 64 + d;
    float edge = 0.f;
    #pragma unroll
    for (int f = 0; f < EF; ++f) edge += aerow[f] * wv[(size_t)f * 64];
    float den = denom[i * 4 + h];
    float v = (nodeagg[((size_t)h * N_NODES + i) * 64 + d] + edge) / den;
    float y = v > 0.f ? v : (__expf(v) - 1.0f);
    out[(size_t)i * 768 + 256 + l * 256 + c] = y;
    cur[(size_t)i * 256 + c] = (_Float16)y;
}

extern "C" void kernel_launch(void* const* d_in, const int* in_sizes, int n_in,
                              void* d_out, int out_size, void* d_ws, size_t ws_size,
                              hipStream_t stream) {
    (void)in_sizes; (void)n_in; (void)out_size; (void)ws_size;
    const float* x  = (const float*)d_in[0];
    const int*   adj= (const int*)d_in[1];
    const float* e  = (const float*)d_in[2];
    const float* Wq = (const float*)d_in[3];
    const float* Wk = (const float*)d_in[4];
    const float* Wv = (const float*)d_in[5];
    const float* a  = (const float*)d_in[6];
    float* out = (float*)d_out;

    char* w = (char*)d_ws;
    size_t off = 0;
    auto alloc = [&](size_t bytes) { void* p = w + off; off += (bytes + 255) & ~(size_t)255; return p; };
    _Float16* cur  = (_Float16*)alloc((size_t)1024 * 256 * 2);        // f16 layer input
    _Float16* wt   = (_Float16*)alloc((size_t)768 * 256 * 2);         // packed B^T weights
    float* qkv     = (float*)alloc((size_t)1024 * 768 * 4);           // [hq|kn|vn] x 4 heads
    float* sq      = (float*)alloc((size_t)4096 * 4);
    float* sk      = (float*)alloc((size_t)4096 * 4);
    _Float16* vnT  = (_Float16*)alloc((size_t)4 * 64 * 1024 * 2);     // vn transposed, f16
    float* cedge   = (float*)alloc((size_t)256 * 4);
    float* se      = (float*)alloc((size_t)8 * 1024 * 1024 * 4);      // 8 edge-score planes
    _Float16* att  = (_Float16*)alloc((size_t)4 * 1024 * 1024 * 2);   // unnormalized att, f16
    float* denom   = (float*)alloc((size_t)4096 * 4);
    float* ae      = (float*)alloc((size_t)4096 * EF * 4);            // att-weighted e
    float* nodeagg = (float*)alloc((size_t)4 * 1024 * 64 * 4);        // att @ vn

    init_kernel<<<1024, 256, 0, stream>>>(x, out, cur);
    cedge_kernel<<<1, 256, 0, stream>>>(Wk, a, cedge);
    edge_scores_kernel<<<4096, 256, 0, stream>>>(e, cedge, se);

    for (int l = 0; l < 2; ++l) {
        pack_w_kernel<<<768, 256, 0, stream>>>(Wq, Wk, Wv, l, wt);
        // qkv = cur (1024x256) @ W (256x768)
        wmma_gemm_kernel<<<dim3(768 / 64, 1024 / 32, 1), 256, 0, stream>>>(
            cur, wt, qkv, 256, 256, 768, 256, 0, 0, 0);
        qkv_post_kernel<<<16, 256, 0, stream>>>(qkv, a, l, sq, sk, vnT);
        attn_kernel<<<dim3(1024, 4), 256, 0, stream>>>(se, adj, e, sq, sk, l, att, denom, ae);
        // nodeagg[h] = att[h] (1024x1024) @ vn[h] (1024x64), batched over 4 heads
        wmma_gemm_kernel<<<dim3(64 / 64, 1024 / 32, 4), 256, 0, stream>>>(
            att, vnT, nodeagg, 1024, 1024, 64, 1024,
            (size_t)1024 * 1024, (size_t)64 * 1024, (size_t)1024 * 64);
        finalize_kernel<<<1024, 256, 0, stream>>>(nodeagg, ae, denom, Wv, l, out, cur);
    }
}
